// LSTM_6751688589711
// MI455X (gfx1250) — matrix operand order repaired and checked by reading, not compiled, not run on hardware
//
#include <hip/hip_runtime.h>
#include <hip/hip_bf16.h>

// ---------------------------------------------------------------------------
// LSTM on MI455X (gfx1250, wave32, WMMA), fully fragment-resident version.
//
//  gates[64,2048] = [x_t ; h] @ [Wx ; Wh]   (K = 1024) per timestep.
//
//  - Weights packed ONCE to bf16 B-fragment layout (4 MB, L2-resident x512).
//  - x packed ONCE to bf16 A-fragment layout for ALL timesteps (32 MB; the
//    64 KB per-step slice streams from HBM/L2).
//  - h kept ONLY as bf16 A-fragments (64 KB double buffer): each step's
//    elementwise phase writes h_new directly in fragment layout, so the step
//    kernel has NO staging loop at all -- the K-loop is pure coalesced
//    global_load_b128 + v_wmma_f32_16x16x32_bf16 (f32 accumulate).
//  - One lean kernel launch per timestep (graph-captured); LDS used only for
//    the 4 KB inter-wave gate-tile exchange.
// ---------------------------------------------------------------------------

typedef __attribute__((ext_vector_type(16))) __bf16 v16bf;
typedef __attribute__((ext_vector_type(8)))  float  v8f;

#define LSTM_N 64
#define LSTM_T 512
#define LSTM_D 512
#define LSTM_H 512
#define G4H    2048            // 4*H
#define KTOT   1024            // D + H
#define KT     32              // KTOT / 32 WMMA K-steps (16 x-part + 16 h-part)

__device__ __forceinline__ unsigned short f2bf(float f) {
    // round-to-nearest-even fp32 -> bf16
    unsigned u = __builtin_bit_cast(unsigned, f);
    unsigned r = u + 0x7FFFu + ((u >> 16) & 1u);
    return (unsigned short)(r >> 16);
}

// A-fragment k_local for (lane, e):  (ISA 7.12.2, 16-bit A 16x32)
__device__ __forceinline__ int a_klocal(int lane, int e) {
    return ((lane & 16) ? 8 : 0) + ((e < 8) ? e : e + 8);
}

// ---------------------------------------------------------------------------
// Pack [Wx ; Wh] into bf16 B-fragment layout:
//   flat = ((colt*KT + kt)*32 + lane)*16 + e
//   col  = colt*16 + (lane&15) ;  k = kt*32 + a_klocal(lane,e)
// ---------------------------------------------------------------------------
__global__ __launch_bounds__(256)
void lstm_pack_w(const float* __restrict__ Wx, const float* __restrict__ Wh,
                 unsigned short* __restrict__ packedW) {
    int idx  = blockIdx.x * 256 + threadIdx.x;       // 0 .. KTOT*G4H-1
    int e    = idx & 15;
    int lane = (idx >> 4) & 31;
    int kt   = (idx >> 9) & (KT - 1);
    int colt = idx >> 14;
    int col  = colt * 16 + (lane & 15);
    int k    = kt * 32 + a_klocal(lane, e);
    float v  = (k < LSTM_D) ? Wx[(size_t)k * G4H + col]
                            : Wh[(size_t)(k - LSTM_D) * G4H + col];
    packedW[idx] = f2bf(v);
}

// ---------------------------------------------------------------------------
// Pack x[N,T,D] into bf16 A-fragment layout for all timesteps:
//   flat = (((t*4 + mt)*16 + kt)*32 + lane)*16 + e
//   row  = lane&15 ; n = mt*16+row ; k = kt*32 + a_klocal(lane,e)
// ---------------------------------------------------------------------------
__global__ __launch_bounds__(256)
void lstm_pack_x(const float* __restrict__ x, unsigned short* __restrict__ xp) {
    size_t idx = (size_t)blockIdx.x * 256 + threadIdx.x; // 0 .. N*T*D-1
    int e    = (int)(idx & 15);
    int lane = (int)(idx >> 4) & 31;
    int kt   = (int)(idx >> 9) & 15;
    int mt   = (int)(idx >> 13) & 3;
    int t    = (int)(idx >> 15);
    int n    = mt * 16 + (lane & 15);
    int k    = kt * 32 + a_klocal(lane, e);
    xp[idx] = f2bf(x[((size_t)n * LSTM_T + t) * LSTM_D + k]);
}

// c = 0, hfrag0 = A-fragment-packed h0
__global__ __launch_bounds__(256)
void lstm_init(const float* __restrict__ h0, unsigned short* __restrict__ hfrag,
               float* __restrict__ c) {
    int idx = blockIdx.x * 256 + threadIdx.x;        // 0 .. N*H-1 (== frag count)
    c[idx] = 0.0f;
    int e    = idx & 15;
    int lane = (idx >> 4) & 31;
    int kt   = (idx >> 9) & 15;
    int mt   = idx >> 13;
    int n    = mt * 16 + (lane & 15);
    int k    = kt * 32 + a_klocal(lane, e);
    hfrag[idx] = f2bf(h0[(size_t)n * LSTM_H + k]);
}

// ---------------------------------------------------------------------------
// One LSTM timestep.  grid = 128 blocks (mt = blockIdx&3, ct = blockIdx>>2),
// 128 threads = 4 waves; wave g owns gate g's 16x16 tile.
// ---------------------------------------------------------------------------
__global__ __launch_bounds__(128)
void lstm_step(const unsigned short* __restrict__ xpack,
               const float* __restrict__ b,
               const unsigned short* __restrict__ packedW,
               const unsigned short* __restrict__ hfrag_prev,
               unsigned short* __restrict__ hfrag_next,
               float* __restrict__ c, float* __restrict__ out, int t) {
    __shared__ float gbuf[4][256];                   // gate tiles (i,f,o,g)

    const int tid  = threadIdx.x;
    const int mt   = blockIdx.x & 3;
    const int ct   = blockIdx.x >> 2;
    const int wave = tid >> 5;                       // gate id
    const int lane = tid & 31;
    const int coltG = wave * (LSTM_H / 16) + ct;     // global col-tile 0..127

    const v16bf* bp  = (const v16bf*)packedW + (size_t)coltG * KT * 32;
    const v16bf* axp = (const v16bf*)xpack + (((size_t)t * 4 + mt) * 16) * 32;
    const v16bf* ahp = (const v16bf*)hfrag_prev + (size_t)mt * 16 * 32;

    v8f acc = {};
#pragma unroll
    for (int kt = 0; kt < 16; ++kt) {                // x-part: B rows = Wx
        v16bf a  = axp[kt * 32 + lane];
        v16bf bb = bp [kt * 32 + lane];
        acc = __builtin_amdgcn_wmma_f32_16x16x32_bf16(
                  false, a, false, bb, (short)0, acc, false, false);
    }
#pragma unroll
    for (int kt = 0; kt < 16; ++kt) {                // h-part: B rows = Wh
        v16bf a  = ahp[kt * 32 + lane];
        v16bf bb = bp [(16 + kt) * 32 + lane];
        acc = __builtin_amdgcn_wmma_f32_16x16x32_bf16(
                  false, a, false, bb, (short)0, acc, false, false);
    }

    // Scatter accumulator to LDS (C layout: VGPR r -> row r + (lane>=16)*8)
    {
        int col   = lane & 15;
        int rbase = (lane & 16) ? 8 : 0;
#pragma unroll
        for (int r = 0; r < 8; ++r)
            gbuf[wave][(rbase + r) * 16 + col] = acc[r];
    }
    __syncthreads();

    // Elementwise LSTM cell update: 256 elements, 2 per thread.
    for (int i = tid; i < 256; i += 128) {
        int row  = i >> 4;
        int col  = i & 15;
        int n    = mt * 16 + row;
        int hcol = ct * 16 + col;

        float iv = gbuf[0][i] + b[0 * LSTM_H + hcol];
        float fv = gbuf[1][i] + b[1 * LSTM_H + hcol];
        float ov = gbuf[2][i] + b[2 * LSTM_H + hcol];
        float gv = gbuf[3][i] + b[3 * LSTM_H + hcol];

        iv = 1.0f / (1.0f + __expf(-iv));
        fv = 1.0f / (1.0f + __expf(-fv));
        ov = 1.0f / (1.0f + __expf(-ov));
        gv = tanhf(gv);

        size_t cidx = (size_t)n * LSTM_H + hcol;
        float cn = fv * c[cidx] + iv * gv;
        c[cidx] = cn;
        float hn = ov * tanhf(cn);

        // h_new -> bf16 A-fragment layout (inverse of a_klocal mapping)
        int kt = hcol >> 5;
        int kl = hcol & 31;
        int hi = (kl >> 3) & 1;
        int e  = (kl & 7) + ((kl & 16) ? 8 : 0);
        int lf = hi * 16 + row;
        hfrag_next[(((size_t)mt * 16 + kt) * 32 + lf) * 16 + e] = f2bf(hn);

        out[((size_t)n * LSTM_T + t) * LSTM_H + hcol] = hn;
    }
}

extern "C" void kernel_launch(void* const* d_in, const int* in_sizes, int n_in,
                              void* d_out, int out_size, void* d_ws, size_t ws_size,
                              hipStream_t stream) {
    (void)in_sizes; (void)n_in; (void)out_size; (void)ws_size;
    const float* x  = (const float*)d_in[0];
    const float* h0 = (const float*)d_in[1];
    const float* Wx = (const float*)d_in[2];
    const float* Wh = (const float*)d_in[3];
    const float* b  = (const float*)d_in[4];
    float* out = (float*)d_out;

    char* ws = (char*)d_ws;
    unsigned short* packedW = (unsigned short*)ws;                    // 4 MiB
    size_t off = (size_t)KTOT * G4H * sizeof(unsigned short);
    unsigned short* xpack = (unsigned short*)(ws + off);              // 32 MiB
    off += (size_t)LSTM_N * LSTM_T * LSTM_D * sizeof(unsigned short);
    float* c = (float*)(ws + off);                                    // 128 KiB
    off += (size_t)LSTM_N * LSTM_H * sizeof(float);
    unsigned short* hA = (unsigned short*)(ws + off);                 // 64 KiB
    off += (size_t)LSTM_N * LSTM_H * sizeof(unsigned short);
    unsigned short* hB = (unsigned short*)(ws + off);                 // 64 KiB

    lstm_pack_w<<<(KTOT * G4H) / 256, 256, 0, stream>>>(Wx, Wh, packedW);
    lstm_pack_x<<<(int)(((size_t)LSTM_N * LSTM_T * LSTM_D) / 256), 256, 0,
                  stream>>>(x, xpack);
    lstm_init<<<(LSTM_N * LSTM_H) / 256, 256, 0, stream>>>(h0, hA, c);

    for (int t = 0; t < LSTM_T; ++t) {
        const unsigned short* hp = (t & 1) ? hB : hA;
        unsigned short*       hn = (t & 1) ? hA : hB;
        lstm_step<<<128, 128, 0, stream>>>(xpack, b, packedW, hp, hn, c, out, t);
    }
}
